// SSMFoundational_91216515432923
// MI455X (gfx1250) — matrix operand order repaired and checked by reading, not compiled, not run on hardware
//
#include <hip/hip_runtime.h>

// ---------------- problem constants ----------------
#define TT   32768
#define RR   256
#define GG   4
#define HH   512
#define PH   256
#define LL   6
#define CTX  8
#define OO   128
#define HC   504   // H - CTX

typedef unsigned short u16;
typedef __attribute__((ext_vector_type(16))) __bf16 v16bf;
typedef __attribute__((ext_vector_type(8)))  float  v8f;
typedef int ssm_v4i __attribute__((vector_size(16)));

#ifndef __has_builtin
#define __has_builtin(x) 0
#endif

// CDNA5 async global->LDS path (ASYNCcnt-tracked), with safe fallback.
#if defined(__gfx1250__) && __has_builtin(__builtin_amdgcn_global_load_async_to_lds_b128)
#define SSM_ASYNC 1
#else
#define SSM_ASYNC 0
#endif

static __device__ inline void ssm_async16(const u16* g, u16* l) {
#if SSM_ASYNC
    __builtin_amdgcn_global_load_async_to_lds_b128(
        (__attribute__((address_space(1))) ssm_v4i*)(void*)g,
        (__attribute__((address_space(3))) ssm_v4i*)(void*)l, 0, 0);
#else
    (void)g; (void)l;
#endif
}
static __device__ inline void ssm_wait_async() {
#if SSM_ASYNC
#if __has_builtin(__builtin_amdgcn_s_wait_asynccnt)
    __builtin_amdgcn_s_wait_asynccnt(0);
#else
    asm volatile("s_wait_asynccnt 0x0" ::: "memory");
#endif
#endif
}

static __device__ inline u16 f2bf(float f) {
    unsigned int u = __float_as_uint(f);
    unsigned int r = (u + 0x7FFFu + ((u >> 16) & 1u)) >> 16;
    return (u16)r;
}
static __device__ inline float bf2f(u16 h) {
    return __uint_as_float(((unsigned int)h) << 16);
}
static __device__ inline float gelu_tanh(float x) {
    return 0.5f * x * (1.f + tanhf(0.7978845608028654f * (x + 0.044715f * x * x * x)));
}

// ---------------- conversion / prep kernels ----------------
__global__ __launch_bounds__(256) void ssm_f2bf_kernel(const float* __restrict__ src,
                                                       u16* __restrict__ dst) {
    size_t i = (size_t)blockIdx.x * 256 + threadIdx.x;
    dst[i] = f2bf(src[i]);
}

__global__ __launch_bounds__(256) void ssm_encw_kernel(const float* __restrict__ enc_w,
                                                       const int* __restrict__ gptr,
                                                       u16* __restrict__ dst) {
    int i = blockIdx.x * 256 + threadIdx.x;     // 512*256 elements
    int n = i >> 8, k = i & 255;
    int g = *gptr;
    float v = (n < HC) ? enc_w[((size_t)g * HC + n) * RR + k] : 0.f;
    dst[i] = f2bf(v);
}

__global__ __launch_bounds__(256) void ssm_decw_kernel(const float* __restrict__ dec_w,
                                                       const int* __restrict__ gptr,
                                                       u16* __restrict__ dst) {
    int i = blockIdx.x * 256 + threadIdx.x;     // 128*512 elements
    int g = *gptr;
    dst[i] = f2bf(dec_w[(size_t)g * (OO * HH) + i]);
}

__global__ __launch_bounds__(256) void ssm_ctx_kernel(const float* __restrict__ ctx_emb,
                                                      const int* __restrict__ gptr,
                                                      float* __restrict__ hF) {
    int i = blockIdx.x * 256 + threadIdx.x;     // T*CTX
    int t = i >> 3, j = i & 7;
    int g = *gptr;
    hF[(size_t)t * HH + HC + j] = ctx_emb[g * CTX + j];
}

// Lbar = exp(Lam*dt); Bbar = ((Lbar-1)/Lam) * (B_re + i B_im), stacked (2Ph x H) bf16
__global__ __launch_bounds__(512) void ssm_prep_kernel(const float* __restrict__ lam_re,
                                                       const float* __restrict__ lam_im,
                                                       const float* __restrict__ logstep,
                                                       const float* __restrict__ Bre,
                                                       const float* __restrict__ Bim,
                                                       float* __restrict__ LbarF,
                                                       u16* __restrict__ BbarW) {
    int p = blockIdx.x;      // Ph
    int h = threadIdx.x;     // H
    __shared__ float sv[2];
    if (h == 0) {
        float lr = lam_re[p], li = lam_im[p];
        float dt = expf(logstep[p]);
        float a = lr * dt, b = li * dt;
        float ea = expf(a);
        float lbr = ea * cosf(b), lbi = ea * sinf(b);
        float nr = lbr - 1.f, ni = lbi;
        float den = lr * lr + li * li;
        sv[0] = (nr * lr + ni * li) / den;
        sv[1] = (ni * lr - nr * li) / den;
        LbarF[2 * p] = lbr;
        LbarF[2 * p + 1] = lbi;
    }
    __syncthreads();
    float sr = sv[0], si = sv[1];
    float br = Bre[(size_t)p * HH + h], bi = Bim[(size_t)p * HH + h];
    BbarW[(size_t)p * HH + h]        = f2bf(sr * br - si * bi);
    BbarW[(size_t)(PH + p) * HH + h] = f2bf(sr * bi + si * br);
}

// Wc row h = [2*C_re[h,:] , -2*C_im[h,:]]  (H x 2Ph) bf16
__global__ __launch_bounds__(256) void ssm_wc_kernel(const float* __restrict__ Cre,
                                                     const float* __restrict__ Cim,
                                                     u16* __restrict__ WcW) {
    int h = blockIdx.x;      // H
    int p = threadIdx.x;     // Ph
    WcW[(size_t)h * (2 * PH) + p]      = f2bf( 2.f * Cre[(size_t)h * PH + p]);
    WcW[(size_t)h * (2 * PH) + PH + p] = f2bf(-2.f * Cim[(size_t)h * PH + p]);
}

// ---------------- layernorm (row of 512) -> bf16 ----------------
__global__ __launch_bounds__(256) void ssm_ln_kernel(const float* __restrict__ h,
                                                     const float* __restrict__ scale,
                                                     const float* __restrict__ bias,
                                                     u16* __restrict__ out) {
    int row = blockIdx.x;
    int tid = threadIdx.x;
    const float2 v = ((const float2*)(h + (size_t)row * HH))[tid];
    float s = v.x + v.y;
    float q = v.x * v.x + v.y * v.y;
    for (int o = 16; o > 0; o >>= 1) {
        s += __shfl_down(s, o);
        q += __shfl_down(q, o);
    }
    __shared__ float ss[8], sq[8];
    int wid = tid >> 5, lane = tid & 31;
    if (lane == 0) { ss[wid] = s; sq[wid] = q; }
    __syncthreads();
    if (tid == 0) {
        float a = 0.f, b = 0.f;
        for (int i = 0; i < 8; i++) { a += ss[i]; b += sq[i]; }
        ss[0] = a; sq[0] = b;
    }
    __syncthreads();
    float mu = ss[0] * (1.f / HH);
    float var = sq[0] * (1.f / HH) - mu * mu;
    float rstd = rsqrtf(var + 1e-6f);
    int n = 2 * tid;
    out[(size_t)row * HH + n]     = f2bf((v.x - mu) * rstd * scale[n] + bias[n]);
    out[(size_t)row * HH + n + 1] = f2bf((v.y - mu) * rstd * scale[n + 1] + bias[n + 1]);
}

// ---------------- chunked complex associative scan (one wave per channel) ----------------
__global__ __launch_bounds__(32) void ssm_scan_kernel(const float* __restrict__ Bu,
                                                      const float* __restrict__ LbarF,
                                                      const float* __restrict__ st_in,
                                                      float* __restrict__ st_out,
                                                      u16* __restrict__ xsBF) {
    int c = blockIdx.x;      // Ph channels
    int lane = threadIdx.x;  // wave32
    const float Ar = LbarF[2 * c], Ai = LbarF[2 * c + 1];
    const float s0r = st_in[2 * c], s0i = st_in[2 * c + 1];
    float cr = 0.f, ci = 0.f;   // carry state
    for (int t0 = 0; t0 < TT; t0 += 32) {
        int t = t0 + lane;
        float br = Bu[(size_t)t * (2 * PH) + c];
        float bi = Bu[(size_t)t * (2 * PH) + PH + c];
        if (t0 == 0 && lane == 0) {          // fold Lbar*state into Bu[0]
            br += Ar * s0r - Ai * s0i;
            bi += Ar * s0i + Ai * s0r;
        }
        float ar = Ar, ai = Ai;
        #pragma unroll
        for (int off = 1; off < 32; off <<= 1) {
            float par = __shfl_up(ar, off);
            float pai = __shfl_up(ai, off);
            float pbr = __shfl_up(br, off);
            float pbi = __shfl_up(bi, off);
            if (lane >= off) {
                float nbr = ar * pbr - ai * pbi + br;
                float nbi = ar * pbi + ai * pbr + bi;
                float nar = ar * par - ai * pai;
                float nai = ar * pai + ai * par;
                br = nbr; bi = nbi; ar = nar; ai = nai;
            }
        }
        float xr = br + ar * cr - ai * ci;
        float xi = bi + ar * ci + ai * cr;
        xsBF[(size_t)t * (2 * PH) + c]      = f2bf(xr);
        xsBF[(size_t)t * (2 * PH) + PH + c] = f2bf(xi);
        cr = __shfl(xr, 31);
        ci = __shfl(xi, 31);
    }
    if (lane == 0) { st_out[2 * c] = cr; st_out[2 * c + 1] = ci; }
}

// ---------------- WMMA bf16 GEMM: out = A(MxK) @ W(NxK)^T  (+fused epilogues) ----------------
// block tile 128x128, 8 waves (4 along M x 2 along N), wave tile 32x64.
// KC=64 with double-buffered LDS (64KB total); async global->LDS prefetch of chunk k+1
// overlapped with WMMA on chunk k (register-prefetch fallback when builtin is absent).
#define BM 128
#define BN 128
#define KC 64

enum { EPI_ENC = 0, EPI_STORE = 1, EPI_CY = 2, EPI_GLU = 3, EPI_DEC = 4 };

union FragU { uint4 u[2]; v16bf v; };

template <int EPI>
__global__ __launch_bounds__(256) void ssm_gemm_kernel(
    const u16* __restrict__ A, const u16* __restrict__ W,
    float* __restrict__ out, int K, int ldo, int Nvalid,
    const float* __restrict__ bias,        // ENC/DEC: group bias base; CY: D; GLU: glu_b
    const float* __restrict__ aux1,        // GLU: y buffer
    const u16* __restrict__ hnbf,          // CY: layernormed h (bf16)
    u16* __restrict__ gybf,                // CY: gelu(y) output (bf16)
    const int* __restrict__ gptr, int gstride) {
    __shared__ __align__(16) u16 lA[2][BM * KC];
    __shared__ __align__(16) u16 lB[2][BN * KC];
    const int tid = threadIdx.x;
    const int m0 = blockIdx.y * BM;
    const int n0 = blockIdx.x * BN;
    const int wid = tid >> 5, lane = tid & 31;
    const int wm = wid >> 1, wn = wid & 1;
    const int half = lane >> 4, lr = lane & 15;

    // staging assignment: thread copies 32 u16 (64B = 4x b128) per tile per chunk
    const int srow = tid >> 1;            // 0..127
    const int scol = (tid & 1) * 32;      // 0 or 32 (KC=64)
    const u16* Abase = A + (size_t)(m0 + srow) * K + scol;
    const u16* Wbase = W + (size_t)(n0 + srow) * K + scol;

    v8f acc[2][4] = {};

    const int nch = K / KC;

    auto compute = [&](int buf) {
        const u16* bA = &lA[buf][0];
        const u16* bB = &lB[buf][0];
        #pragma unroll
        for (int ks = 0; ks < KC; ks += 32) {
            FragU a[2], b[4];
            // A frag 16x32: lane<16 row=lr holds K {k..k+7, k+16..k+23}; lane>=16 +8
            #pragma unroll
            for (int mi = 0; mi < 2; mi++) {
                const u16* base = bA + (size_t)(wm * 32 + mi * 16 + lr) * KC + ks + half * 8;
                a[mi].u[0] = *(const uint4*)(base);
                a[mi].u[1] = *(const uint4*)(base + 16);
            }
            // B frag 32x16 (=W^T): lane<16 N=lr K {k..k+7, k+16..k+23}; lane>=16 +8
            #pragma unroll
            for (int ni = 0; ni < 4; ni++) {
                const u16* base = bB + (size_t)(wn * 64 + ni * 16 + lr) * KC + ks + half * 8;
                b[ni].u[0] = *(const uint4*)(base);
                b[ni].u[1] = *(const uint4*)(base + 16);
            }
            #pragma unroll
            for (int mi = 0; mi < 2; mi++)
                #pragma unroll
                for (int ni = 0; ni < 4; ni++)
                    acc[mi][ni] = __builtin_amdgcn_wmma_f32_16x16x32_bf16(
                        false, a[mi].v, false, b[ni].v, (short)0, acc[mi][ni], false, false);
        }
    };

#if SSM_ASYNC
    auto stage_async = [&](int c, int buf) {
        const u16* ga = Abase + (size_t)c * KC;
        const u16* gw = Wbase + (size_t)c * KC;
        u16* sa = &lA[buf][srow * KC + scol];
        u16* sb = &lB[buf][srow * KC + scol];
        #pragma unroll
        for (int q = 0; q < 4; q++) ssm_async16(ga + q * 8, sa + q * 8);
        #pragma unroll
        for (int q = 0; q < 4; q++) ssm_async16(gw + q * 8, sb + q * 8);
    };
    stage_async(0, 0);
    ssm_wait_async();
    __syncthreads();
    for (int c = 0; c < nch; ++c) {
        int buf = c & 1;
        if (c + 1 < nch) stage_async(c + 1, buf ^ 1);   // prefetch next chunk into other buffer
        compute(buf);
        ssm_wait_async();
        __syncthreads();
    }
#else
    uint4 ra[4], rb[4];
    auto load_regs = [&](int c) {
        const uint4* pa = (const uint4*)(Abase + (size_t)c * KC);
        const uint4* pw = (const uint4*)(Wbase + (size_t)c * KC);
        #pragma unroll
        for (int q = 0; q < 4; q++) ra[q] = pa[q];
        #pragma unroll
        for (int q = 0; q < 4; q++) rb[q] = pw[q];
    };
    auto store_lds = [&](int buf) {
        uint4* sa = (uint4*)(&lA[buf][srow * KC + scol]);
        uint4* sb = (uint4*)(&lB[buf][srow * KC + scol]);
        #pragma unroll
        for (int q = 0; q < 4; q++) sa[q] = ra[q];
        #pragma unroll
        for (int q = 0; q < 4; q++) sb[q] = rb[q];
    };
    load_regs(0);
    store_lds(0);
    __syncthreads();
    for (int c = 0; c < nch; ++c) {
        int buf = c & 1;
        bool pf = (c + 1 < nch);
        if (pf) load_regs(c + 1);   // global loads in flight during WMMA
        compute(buf);
        if (pf) store_lds(buf ^ 1);
        __syncthreads();
    }
#endif

    int g = 0;
    if (EPI == EPI_ENC || EPI == EPI_DEC) g = *gptr;

    #pragma unroll
    for (int mi = 0; mi < 2; mi++) {
        #pragma unroll
        for (int ni = 0; ni < 4; ni++) {
            #pragma unroll
            for (int r = 0; r < 8; r++) {
                int M = m0 + wm * 32 + mi * 16 + half * 8 + r;
                int N = n0 + wn * 64 + ni * 16 + lr;
                if (N >= Nvalid) continue;
                float v = acc[mi][ni][r];
                size_t idx = (size_t)M * ldo + N;
                if constexpr (EPI == EPI_ENC) {
                    out[idx] = v + bias[(size_t)g * gstride + N];
                } else if constexpr (EPI == EPI_STORE) {
                    out[idx] = v;
                } else if constexpr (EPI == EPI_CY) {
                    float y = v + bias[N] * bf2f(hnbf[idx]);
                    out[idx] = y;
                    gybf[idx] = f2bf(gelu_tanh(y));
                } else if constexpr (EPI == EPI_GLU) {
                    float z = v + bias[N];
                    out[idx] = out[idx] + aux1[idx] * (1.f / (1.f + expf(-z)));
                } else {  // EPI_DEC
                    out[idx] = v + bias[(size_t)g * gstride + N];
                }
            }
        }
    }
}

// ---------------- host launcher ----------------
extern "C" void kernel_launch(void* const* d_in, const int* in_sizes, int n_in,
                              void* d_out, int out_size, void* d_ws, size_t ws_size,
                              hipStream_t stream) {
    (void)in_sizes; (void)n_in; (void)out_size; (void)ws_size;
    const float* x        = (const float*)d_in[0];
    const float* state    = (const float*)d_in[1];
    const float* enc_w    = (const float*)d_in[2];
    const float* enc_b    = (const float*)d_in[3];
    const float* ctx_emb  = (const float*)d_in[4];
    const float* ln_scale = (const float*)d_in[5];
    const float* ln_bias  = (const float*)d_in[6];
    const float* Lam_re   = (const float*)d_in[7];
    const float* Lam_im   = (const float*)d_in[8];
    const float* B_re     = (const float*)d_in[9];
    const float* B_im     = (const float*)d_in[10];
    const float* C_re     = (const float*)d_in[11];
    const float* C_im     = (const float*)d_in[12];
    const float* Dd       = (const float*)d_in[13];
    const float* log_step = (const float*)d_in[14];
    const float* glu_w    = (const float*)d_in[15];
    const float* glu_b    = (const float*)d_in[16];
    const float* dec_w    = (const float*)d_in[17];
    const float* dec_b    = (const float*)d_in[18];
    const int*   gptr     = (const int*)d_in[19];
    float* outF = (float*)d_out;

    char* ws = (char*)d_ws;
    size_t off = 0;
    auto alloc = [&](size_t bytes) -> char* {
        char* p = ws + off;
        off += (bytes + 255) & ~(size_t)255;
        return p;
    };
    u16*   xBF   = (u16*)alloc((size_t)TT * RR * 2);
    float* hF    = (float*)alloc((size_t)TT * HH * 4);
    u16*   hnBF  = (u16*)alloc((size_t)TT * HH * 2);
    float* BuF   = (float*)alloc((size_t)TT * 2 * PH * 4);  // reused as yF
    u16*   xsBF  = (u16*)alloc((size_t)TT * 2 * PH * 2);
    u16*   gyBF  = (u16*)alloc((size_t)TT * HH * 2);
    u16*   encW  = (u16*)alloc(512 * 256 * 2);
    u16*   BbarW = (u16*)alloc(512 * 512 * 2);
    u16*   WcW   = (u16*)alloc(512 * 512 * 2);
    u16*   gluW  = (u16*)alloc(512 * 512 * 2);
    u16*   decW  = (u16*)alloc(128 * 512 * 2);
    float* LbarF = (float*)alloc(PH * 2 * 4);
    float* stF   = (float*)alloc(PH * 2 * 4);
    float* yF = BuF;

    // 1) convert inputs / group weights to bf16
    ssm_f2bf_kernel<<<(TT * RR) / 256, 256, 0, stream>>>(x, xBF);
    ssm_encw_kernel<<<(512 * 256) / 256, 256, 0, stream>>>(enc_w, gptr, encW);
    ssm_decw_kernel<<<(128 * 512) / 256, 256, 0, stream>>>(dec_w, gptr, decW);

    // 2) encoder GEMM: h[:, :504] = x @ enc_w[g]^T + enc_b[g]; then ctx fill
    ssm_gemm_kernel<EPI_ENC><<<dim3(512 / BN, TT / BM), 256, 0, stream>>>(
        xBF, encW, hF, RR, HH, HC, enc_b, nullptr, nullptr, nullptr, gptr, HC);
    ssm_ctx_kernel<<<(TT * CTX) / 256, 256, 0, stream>>>(ctx_emb, gptr, hF);

    // 3) layers
    for (int i = 0; i < LL; i++) {
        ssm_prep_kernel<<<PH, HH, 0, stream>>>(
            Lam_re + (size_t)i * PH, Lam_im + (size_t)i * PH, log_step + (size_t)i * PH,
            B_re + (size_t)i * PH * HH, B_im + (size_t)i * PH * HH, LbarF, BbarW);
        ssm_wc_kernel<<<HH, PH, 0, stream>>>(
            C_re + (size_t)i * HH * PH, C_im + (size_t)i * HH * PH, WcW);
        ssm_f2bf_kernel<<<(HH * HH) / 256, 256, 0, stream>>>(glu_w + (size_t)i * HH * HH, gluW);

        ssm_ln_kernel<<<TT, 256, 0, stream>>>(hF, ln_scale + (size_t)i * HH,
                                              ln_bias + (size_t)i * HH, hnBF);
        // Bu = hn @ [Bbar_re; Bbar_im]^T  (T x 2Ph, f32)
        ssm_gemm_kernel<EPI_STORE><<<dim3((2 * PH) / BN, TT / BM), 256, 0, stream>>>(
            hnBF, BbarW, BuF, HH, 2 * PH, 2 * PH, nullptr, nullptr, nullptr, nullptr, nullptr, 0);
        // associative scan over T for each channel
        ssm_scan_kernel<<<PH, 32, 0, stream>>>(BuF, LbarF, (i == 0) ? state : stF, stF, xsBF);
        // y = [xs_re|xs_im] @ Wc^T + D*hn ; gy = gelu(y) bf16   (yF aliases BuF, now dead)
        ssm_gemm_kernel<EPI_CY><<<dim3(HH / BN, TT / BM), 256, 0, stream>>>(
            xsBF, WcW, yF, 2 * PH, HH, HH, Dd + (size_t)i * HH, nullptr, hnBF, gyBF, nullptr, 0);
        // h += y * sigmoid(gy @ glu_w^T + glu_b)   (in-place residual)
        ssm_gemm_kernel<EPI_GLU><<<dim3(HH / BN, TT / BM), 256, 0, stream>>>(
            gyBF, gluW, hF, HH, HH, HH, glu_b + (size_t)i * HH, yF, nullptr, nullptr, nullptr, 0);
    }

    // 4) decoder: out = h @ dec_w[g]^T + dec_b[g]
    ssm_f2bf_kernel<<<(TT * HH) / 256, 256, 0, stream>>>(hF, hnBF);
    ssm_gemm_kernel<EPI_DEC><<<dim3(OO / BN + ((OO % BN) ? 1 : 0), TT / BM), 256, 0, stream>>>(
        hnBF, decW, outF, HH, OO, OO, dec_b, nullptr, nullptr, nullptr, gptr, OO);
}
